// Attention_40991167873464
// MI455X (gfx1250) — matrix operand order repaired and checked by reading, not compiled, not run on hardware
//
#include <hip/hip_runtime.h>
#include <hip/hip_bf16.h>
#include <math.h>

typedef __attribute__((ext_vector_type(16))) __bf16 v16bf;
typedef __attribute__((ext_vector_type(8)))  __bf16 v8bf;
typedef __attribute__((ext_vector_type(8)))  float  v8f;

#define B_SZ 32
#define T_SZ 2048
#define DIM  1024

__device__ __forceinline__ unsigned short f2bf(float f) {
    union { float f; unsigned u; } x; x.f = f;
    unsigned u = x.u;
    unsigned r = u + 0x7FFFu + ((u >> 16) & 1u);   // round-to-nearest-even
    return (unsigned short)(r >> 16);
}

// ---------------------------------------------------------------------------
// W_K f32 [1024,1024] -> bf16, packed in WMMA B-operand tile layout:
// tile (nt, kcIdx) of shape K=32 x N=16 stored as 512 contiguous bf16:
//   lane L (0..31), element e (0..15):  K = kc + (L>=16 ? 8 : 0) + (e<8 ? e : 16 + e - 8),  N = nt*16 + (L&15)
// ---------------------------------------------------------------------------
__global__ void k_prep_wk(const float* __restrict__ WK, unsigned short* __restrict__ wkt) {
    int idx = blockIdx.x * 256 + threadIdx.x;          // < 1048576
    int tileIdx = idx >> 9;
    int within  = idx & 511;
    int lane = within >> 4;
    int e    = within & 15;
    int nt    = tileIdx >> 5;
    int kcIdx = tileIdx & 31;
    int k = kcIdx * 32 + ((lane >= 16) ? 8 : 0) + (e < 8 ? e : 16 + e - 8);
    int a = nt * 16 + (lane & 15);
    wkt[idx] = f2bf(WK[k * DIM + a]);
}

// ---------------------------------------------------------------------------
// Y[32,1024] = X[32,1024] @ W[1024,1024] + bias (+ bias2 if non-null)
// Used for att2pb = query@W_Q + b_Q + b_K  and  awe = kbar@W_V + b_V.
// ---------------------------------------------------------------------------
__global__ void k_gemm32(const float* __restrict__ X, const float* __restrict__ W,
                         const float* __restrict__ bias, const float* __restrict__ bias2,
                         float* __restrict__ Y) {
    __shared__ float q_s[32][64];
    int tid = threadIdx.x;
    int a = blockIdx.x * 64 + (tid & 63);
    int g = tid >> 6;                                   // b-group: rows g*8 .. g*8+7
    float acc[8];
#pragma unroll
    for (int j = 0; j < 8; ++j) acc[j] = 0.f;
    for (int kc = 0; kc < DIM; kc += 64) {
        __syncthreads();
        for (int i = tid; i < 2048; i += 256) {
            int bb = i >> 6, kk = i & 63;
            q_s[bb][kk] = X[bb * DIM + kc + kk];
        }
        __syncthreads();
        for (int kk = 0; kk < 64; ++kk) {
            float w = W[(kc + kk) * DIM + a];
#pragma unroll
            for (int j = 0; j < 8; ++j) acc[j] = fmaf(q_s[g * 8 + j][kk], w, acc[j]);
        }
    }
    float bsum = bias[a] + (bias2 ? bias2[a] : 0.f);
#pragma unroll
    for (int j = 0; j < 8; ++j) Y[(g * 8 + j) * DIM + a] = acc[j] + bsum;
}

// ---------------------------------------------------------------------------
// scores[b,t] = relu(key[b,t,:]@W_K + att2pb[b,:]) . W_f + b_f
// Block: 128 rows of key (one b), 8 waves x 16 rows. bf16 WMMA 16x16x32.
// LDS: key tile as bf16, row stride 1032 (pad 8) to break bank conflicts.
// N-blocking = 8 tiles per pass: 8 independent accumulator chains, A-fragment
// LDS reads amortized over 8 WMMAs each.
// ---------------------------------------------------------------------------
#define ROWSTRIDE 1032
#define NT_BLK 8

__global__ void __launch_bounds__(256, 1) k_scores(
        const float* __restrict__ key, const unsigned short* __restrict__ wkt,
        const float* __restrict__ att2pb, const float* __restrict__ Wf,
        const float* __restrict__ bf, float* __restrict__ scores) {
    extern __shared__ __align__(16) unsigned short sm[];   // [128][ROWSTRIDE] bf16
    const int tid = threadIdx.x;
    const int b  = blockIdx.x >> 4;
    const int t0 = (blockIdx.x & 15) * 128;

    // ---- Phase 1: stage key tile -> LDS (f32 -> bf16) ----
    const float4* kg = (const float4*)(key + (size_t)(b * T_SZ + t0) * DIM);
    for (int i = tid; i < 32768; i += 256) {               // 128 rows * 256 float4
        int row = i >> 8;
        int c4  = i & 255;
        float4 v = kg[row * 256 + c4];
        unsigned lo = (unsigned)f2bf(v.x) | ((unsigned)f2bf(v.y) << 16);
        unsigned hi = (unsigned)f2bf(v.z) | ((unsigned)f2bf(v.w) << 16);
        *(uint2*)(sm + row * ROWSTRIDE + c4 * 4) = make_uint2(lo, hi);
    }
    __syncthreads();

    // ---- Phase 2: WMMA GEMM + fused relu/W_f epilogue ----
    const int wave = tid >> 5;
    const int lane = tid & 31;
    const int hi16 = lane >> 4;
    const int col  = lane & 15;
    // A-operand per ISA 16-bit layout: lanes<16: K in [0..7]+[16..23]; lanes>=16: +8
    const unsigned short* arow =
        sm + (wave * 16 + col) * ROWSTRIDE + (hi16 ? 8 : 0);

    float sacc[8];
#pragma unroll
    for (int i = 0; i < 8; ++i) sacc[i] = 0.f;

    for (int ng = 0; ng < 64 / NT_BLK; ++ng) {
        v8f c[NT_BLK];
#pragma unroll
        for (int j = 0; j < NT_BLK; ++j) c[j] = {};
        for (int kc = 0; kc < DIM; kc += 32) {
            union { v16bf v; v8bf h[2]; } A;
            A.h[0] = *(const v8bf*)(arow + kc);
            A.h[1] = *(const v8bf*)(arow + kc + 16);
            const int kcIdx = kc >> 5;
#pragma unroll
            for (int j = 0; j < NT_BLK; ++j) {
                const int nt = ng * NT_BLK + j;
                const unsigned short* bp = wkt + (((nt << 5) + kcIdx) << 9) + lane * 16;
                union { v16bf v; v8bf h[2]; } Bm;
                Bm.h[0] = *(const v8bf*)(bp);
                Bm.h[1] = *(const v8bf*)(bp + 8);
                c[j] = __builtin_amdgcn_wmma_f32_16x16x32_bf16(
                           false, A.v, false, Bm.v, (short)0, c[j], false, false);
            }
        }
        // Epilogue: C layout: VGPR i, lane -> M = i + 8*hi16, N = col
#pragma unroll
        for (int j = 0; j < NT_BLK; ++j) {
            const int n = (ng * NT_BLK + j) * 16 + col;
            const float add = att2pb[b * DIM + n];
            const float wf  = Wf[n];
#pragma unroll
            for (int i = 0; i < 8; ++i) {
                float v = c[j][i] + add;
                v = v > 0.f ? v : 0.f;
                sacc[i] = fmaf(v, wf, sacc[i]);
            }
        }
    }

    // Reduce over the 16 N-lanes of each half-wave (masks < 16 stay in-half)
#pragma unroll
    for (int m = 1; m < 16; m <<= 1) {
#pragma unroll
        for (int i = 0; i < 8; ++i) sacc[i] += __shfl_xor(sacc[i], m, 32);
    }
    if (col == 0) {
        const float bfv = bf[0];
#pragma unroll
        for (int i = 0; i < 8; ++i)
            scores[b * T_SZ + t0 + wave * 16 + hi16 * 8 + i] = sacc[i] + bfv;
    }
}

// ---------------------------------------------------------------------------
__global__ void k_softmax(const float* __restrict__ scores,
                          float* __restrict__ alpha_ws, float* __restrict__ alpha_out) {
    __shared__ float red[256];
    const int b = blockIdx.x, tid = threadIdx.x;
    const float* s = scores + b * T_SZ;
    float v[8];
    float m = -INFINITY;
#pragma unroll
    for (int i = 0; i < 8; ++i) { v[i] = s[tid + i * 256]; m = fmaxf(m, v[i]); }
    red[tid] = m; __syncthreads();
    for (int st = 128; st > 0; st >>= 1) {
        if (tid < st) red[tid] = fmaxf(red[tid], red[tid + st]);
        __syncthreads();
    }
    m = red[0]; __syncthreads();
    float sum = 0.f;
#pragma unroll
    for (int i = 0; i < 8; ++i) { v[i] = __expf(v[i] - m); sum += v[i]; }
    red[tid] = sum; __syncthreads();
    for (int st = 128; st > 0; st >>= 1) {
        if (tid < st) red[tid] += red[tid + st];
        __syncthreads();
    }
    const float inv = 1.f / red[0];
#pragma unroll
    for (int i = 0; i < 8; ++i) {
        float a = v[i] * inv;
        alpha_ws[b * T_SZ + tid + i * 256]  = a;
        alpha_out[b * T_SZ + tid + i * 256] = a;
    }
}

// kbar[b,k] = sum_t alpha[b,t] * key[b,t,k]
__global__ void k_kbar(const float* __restrict__ key, const float* __restrict__ alpha,
                       float* __restrict__ kbar) {
    const int b   = blockIdx.x >> 2;
    const int col = (blockIdx.x & 3) * 256 + threadIdx.x;
    const float* kp = key + (size_t)b * T_SZ * DIM + col;
    const float* ap = alpha + b * T_SZ;
    float acc = 0.f;
    for (int t = 0; t < T_SZ; t += 4) {
        acc = fmaf(ap[t],     kp[(size_t)(t)     * DIM], acc);
        acc = fmaf(ap[t + 1], kp[(size_t)(t + 1) * DIM], acc);
        acc = fmaf(ap[t + 2], kp[(size_t)(t + 2) * DIM], acc);
        acc = fmaf(ap[t + 3], kp[(size_t)(t + 3) * DIM], acc);
    }
    kbar[b * DIM + col] = acc;
}

// ---------------------------------------------------------------------------
extern "C" void kernel_launch(void* const* d_in, const int* in_sizes, int n_in,
                              void* d_out, int out_size, void* d_ws, size_t ws_size,
                              hipStream_t stream) {
    const float* key   = (const float*)d_in[0];
    const float* query = (const float*)d_in[1];
    const float* W_K   = (const float*)d_in[2];
    const float* b_K   = (const float*)d_in[3];
    const float* W_Q   = (const float*)d_in[4];
    const float* b_Q   = (const float*)d_in[5];
    const float* W_V   = (const float*)d_in[6];
    const float* b_V   = (const float*)d_in[7];
    const float* W_f   = (const float*)d_in[8];
    const float* b_f   = (const float*)d_in[9];

    float* awe_out   = (float*)d_out;                       // [32,1024]
    float* alpha_out = (float*)d_out + B_SZ * DIM;          // [32,2048]

    char* ws = (char*)d_ws;
    unsigned short* wkt = (unsigned short*)(ws);            // 2 MB  bf16 tiled W_K
    float* att2pb = (float*)(ws + (1u << 21));              // 128 KB
    float* scores = (float*)(ws + (1u << 21) + (1u << 17)); // 256 KB
    float* alpha  = (float*)(ws + (1u << 21) + (1u << 17) + (1u << 18));
    float* kbar   = (float*)(ws + (1u << 21) + (1u << 17) + (1u << 18) + (1u << 18));

    k_prep_wk<<<4096, 256, 0, stream>>>(W_K, wkt);
    k_gemm32<<<16, 256, 0, stream>>>(query, W_Q, b_Q, b_K, att2pb);
    k_scores<<<512, 256, 128 * ROWSTRIDE * 2, stream>>>(key, wkt, att2pb, W_f, b_f, scores);
    k_softmax<<<32, 256, 0, stream>>>(scores, alpha, alpha_out);
    k_kbar<<<128, 256, 0, stream>>>(key, alpha, kbar);
    k_gemm32<<<16, 256, 0, stream>>>(kbar, W_V, b_V, nullptr, awe_out);
}
